// SubgraphEncoder_13846974562983
// MI455X (gfx1250) — matrix operand order repaired and checked by reading, not compiled, not run on hardware
//
#include <hip/hip_runtime.h>

// ---------------------------------------------------------------------------
// SubgraphEncoder on MI455X (gfx1250, wave32):
//   deg/dinv -> swizzle weights to bf16 WMMA-fragment layout ->
//   WMMA GEMM1 -> self-loop+bias init -> edge scatter (f32 global atomics) ->
//   WMMA GEMM2 (ReLU fused on A) -> init -> scatter -> WMMA GEMM3 (+bias)
// Edge aggregation is L2-resident (feature table 25.6 MB << 192 MB L2).
// ---------------------------------------------------------------------------

#define F_IN 256
#define HID  128
#define SUBV 50

typedef __bf16 bf16_t;
typedef bf16_t v16bf __attribute__((ext_vector_type(16)));
typedef float  v8f   __attribute__((ext_vector_type(8)));

// ---------------- degree / normalization ----------------------------------

__global__ void k_set_deg(float* __restrict__ deg, int n) {
  int i = blockIdx.x * blockDim.x + threadIdx.x;
  if (i < n) deg[i] = 1.0f;  // self-loop contributes 1 to every degree
}

__global__ void k_count_deg(const int* __restrict__ dst, float* __restrict__ deg, int e) {
  int i = blockIdx.x * blockDim.x + threadIdx.x;
  if (i < e)
    __hip_atomic_fetch_add(&deg[dst[i]], 1.0f, __ATOMIC_RELAXED,
                           __HIP_MEMORY_SCOPE_AGENT);
}

__global__ void k_rsqrt(const float* __restrict__ deg, float* __restrict__ dinv, int n) {
  int i = blockIdx.x * blockDim.x + threadIdx.x;
  if (i < n) dinv[i] = rsqrtf(deg[i]);  // deg >= 1 always (self-loop)
}

// ---------------- weight pre-swizzle: f32 [K][128] -> bf16 fragment layout -
// Bsw flat index: (((kk/32)*8 + ntile)*32 + lane)*16 + e  with
//   lane = (hi,l16), K = kk + 8*hi + e + (e>=8 ? 8 : 0), N = ntile*16 + l16.
// Each lane's 16 bf16 for one (k-tile, n-tile) are contiguous (32B aligned).

__global__ void k_swizzle_b(const float* __restrict__ B, bf16_t* __restrict__ Bsw,
                            int total /* = K*128 */) {
  int i = blockIdx.x * blockDim.x + threadIdx.x;
  if (i >= total) return;
  int e    = i & 15;
  int lane = (i >> 4) & 31;
  int nt   = (i >> 9) & 7;
  int t    = i >> 12;
  int hi   = lane >> 4;
  int l16  = lane & 15;
  int k    = t * 32 + 8 * hi + e + (e >= 8 ? 8 : 0);
  int n    = nt * 16 + l16;
  Bsw[i] = (bf16_t)B[(size_t)k * HID + n];
}

// ---------------- WMMA GEMM: C[M,128] = A[M,K] @ B[K,128] (+bias) ----------
// Block = 256 threads = 8 waves; blockIdx.x = 16-row M tile; wave w owns
// columns [16w, 16w+16). K stepped by 32 with v_wmma_f32_16x16x32_bf16.
// A fragment (16-bit A 16x32 ISA layout): lane(l16,hi):
//   e<8  -> K = kk + 8*hi + e        (8 contiguous floats at kk+8*hi)
//   e>=8 -> K = kk + 16 + 8*hi + e-8 (8 contiguous floats at kk+16+8*hi)
// B fragment comes pre-swizzled: one contiguous 32B load per lane.
// C/D: VGPR r -> row m0 + r + 8*hi, col n0 + l16.
// Tail M handled by clamping the A row (no divergent EXEC save/restore);
// stores stay guarded.

template <bool RELU>
__global__ void __launch_bounds__(256)
k_wmma_gemm(const float* __restrict__ A, const bf16_t* __restrict__ Bsw,
            float* __restrict__ C, int M, int K, const float* __restrict__ bias) {
  const int lane = threadIdx.x & 31;
  const int wave = threadIdx.x >> 5;  // 0..7 -> 16-column N tile
  const int hi   = lane >> 4;
  const int l16  = lane & 15;
  const int m0   = blockIdx.x * 16;
  const int n0   = wave * 16;

  const int mA = min(m0 + l16, M - 1);          // clamped row (tail-safe)
  const float*  pa_row  = A + (size_t)mA * K;
  const bf16_t* pb_base = Bsw + ((size_t)wave * 32 + lane) * 16;

  v8f c;
  const float cb = bias ? bias[n0 + l16] : 0.0f;
#pragma unroll
  for (int r = 0; r < 8; ++r) c[r] = cb;

#pragma unroll 2
  for (int kk = 0; kk < K; kk += 32) {
    // ---- A fragment: four float4 loads, fp32 -> bf16 in registers
    const float* pa = pa_row + kk + 8 * hi;
    float4 q0 = *(const float4*)(pa);
    float4 q1 = *(const float4*)(pa + 4);
    float4 q2 = *(const float4*)(pa + 16);
    float4 q3 = *(const float4*)(pa + 20);
    float af[16] = {q0.x, q0.y, q0.z, q0.w, q1.x, q1.y, q1.z, q1.w,
                    q2.x, q2.y, q2.z, q2.w, q3.x, q3.y, q3.z, q3.w};
    if (RELU) {
#pragma unroll
      for (int e = 0; e < 16; ++e) af[e] = fmaxf(af[e], 0.0f);
    }
    v16bf a;
#pragma unroll
    for (int e = 0; e < 16; ++e) a[e] = (bf16_t)af[e];

    // ---- B fragment: single contiguous 32B load (pre-swizzled bf16)
    v16bf b = *(const v16bf*)(pb_base + (size_t)(kk >> 5) * 4096);

    // D = A*B + C, f32 accumulate (uniform flow; EXEC all-ones here)
    c = __builtin_amdgcn_wmma_f32_16x16x32_bf16(false, a, false, b,
                                                (short)0, c, false, false);
  }

#pragma unroll
  for (int r = 0; r < 8; ++r) {
    int row = m0 + r + 8 * hi;
    if (row < M) C[(size_t)row * HID + (n0 + l16)] = c[r];
  }
}

// ---------------- accumulator init: self-loop term + bias ------------------
// acc[i,f] = t[i,f] * dinv[i]^2 + bias[f]  (fully initializes acc -> no memset)

__global__ void k_init_acc(const float* __restrict__ t, const float* __restrict__ dinv,
                           const float* __restrict__ bias, float* __restrict__ acc,
                           int total) {
  int i = blockIdx.x * blockDim.x + threadIdx.x;
  if (i >= total) return;
  int row = i >> 7;  // /128
  int col = i & 127;
  float di = dinv[row];
  acc[i] = t[i] * di * di + bias[col];
}

// ---------------- edge scatter: one wave32 per edge ------------------------
// Edge id is wave-uniform -> readfirstlane forces the edge-index and dinv
// loads onto the scalar (SMEM) path. Each lane reads a float4 of the src row
// (128 floats / 32 lanes), scales by dinv[src]*dinv[dst], scatter-adds with
// f32 global atomics (all L2-resident).

__global__ void __launch_bounds__(256)
k_edge_scatter(const float* __restrict__ t, float* __restrict__ acc,
               const int* __restrict__ src_a, const int* __restrict__ dst_a,
               const float* __restrict__ dinv, int e_cnt) {
  int gwarp = (blockIdx.x * blockDim.x + threadIdx.x) >> 5;
  int lane  = threadIdx.x & 31;
  if (gwarp >= e_cnt) return;                       // whole-wave uniform exit
  int ew = __builtin_amdgcn_readfirstlane(gwarp);   // force SGPR indexing
  int s = src_a[ew];
  int d = dst_a[ew];
  float w = dinv[s] * dinv[d];
  float4 v = ((const float4*)(t + (size_t)s * HID))[lane];
  float* pd = acc + (size_t)d * HID + lane * 4;
  __hip_atomic_fetch_add(pd + 0, v.x * w, __ATOMIC_RELAXED, __HIP_MEMORY_SCOPE_AGENT);
  __hip_atomic_fetch_add(pd + 1, v.y * w, __ATOMIC_RELAXED, __HIP_MEMORY_SCOPE_AGENT);
  __hip_atomic_fetch_add(pd + 2, v.z * w, __ATOMIC_RELAXED, __HIP_MEMORY_SCOPE_AGENT);
  __hip_atomic_fetch_add(pd + 3, v.w * w, __ATOMIC_RELAXED, __HIP_MEMORY_SCOPE_AGENT);
}

// ---------------------------------------------------------------------------

extern "C" void kernel_launch(void* const* d_in, const int* in_sizes, int n_in,
                              void* d_out, int out_size, void* d_ws, size_t ws_size,
                              hipStream_t stream) {
  const float* x  = (const float*)d_in[0];
  const float* W1 = (const float*)d_in[1];
  const float* b1 = (const float*)d_in[2];
  const float* W2 = (const float*)d_in[3];
  const float* b2 = (const float*)d_in[4];
  const float* Wl = (const float*)d_in[5];
  const float* bl = (const float*)d_in[6];
  const int*   ei = (const int*)d_in[7];

  const int N  = in_sizes[0] / F_IN;   // 50000
  const int E  = in_sizes[7] / 2;      // 1.6M
  const int M3 = N / SUBV;             // 1000
  const int K3 = SUBV * HID;           // 6400

  const int* src = ei;      // edge_index[0, :]
  const int* dst = ei + E;  // edge_index[1, :]

  // workspace layout: deg[N] dinv[N] t1[N*128] acc1[N*128] t2[N*128] (f32)
  // then bf16 swizzled weights W1sw[256*128] W2sw[128*128] Wlsw[6400*128]
  float* ws   = (float*)d_ws;
  float* deg  = ws;
  float* dinv = deg + N;
  float* t1   = dinv + N;
  float* acc1 = t1 + (size_t)N * HID;
  float* t2   = acc1 + (size_t)N * HID;
  bf16_t* W1sw = (bf16_t*)(t2 + (size_t)N * HID);
  bf16_t* W2sw = W1sw + (size_t)F_IN * HID;
  bf16_t* Wlsw = W2sw + (size_t)HID * HID;

  float* out = (float*)d_out;            // [M3, HID]
  float* emb = out + (size_t)M3 * HID;   // [N, HID] second output, in place

  const int thr = 256;

  // 1) symmetric normalization
  k_set_deg  <<<(N + thr - 1) / thr, thr, 0, stream>>>(deg, N);
  k_count_deg<<<(E + thr - 1) / thr, thr, 0, stream>>>(dst, deg, E);
  k_rsqrt    <<<(N + thr - 1) / thr, thr, 0, stream>>>(deg, dinv, N);

  // 2) weight pre-swizzle (tiny: 0.87M elements total)
  k_swizzle_b<<<(F_IN * HID + thr - 1) / thr, thr, 0, stream>>>(W1, W1sw, F_IN * HID);
  k_swizzle_b<<<(HID * HID + thr - 1) / thr, thr, 0, stream>>>(W2, W2sw, HID * HID);
  k_swizzle_b<<<(K3 * HID + thr - 1) / thr, thr, 0, stream>>>(Wl, Wlsw, K3 * HID);

  // 3) layer 1: t1 = x @ W1 ; acc1 = selfloop + b1 ; scatter edges
  k_wmma_gemm<false><<<(N + 15) / 16, thr, 0, stream>>>(x, W1sw, t1, N, F_IN, nullptr);
  k_init_acc<<<(N * HID + thr - 1) / thr, thr, 0, stream>>>(t1, dinv, b1, acc1, N * HID);
  k_edge_scatter<<<(E * 32 + thr - 1) / thr, thr, 0, stream>>>(t1, acc1, src, dst, dinv, E);

  // 4) layer 2: t2 = relu(acc1) @ W2 ; emb = selfloop + b2 ; scatter edges
  k_wmma_gemm<true><<<(N + 15) / 16, thr, 0, stream>>>(acc1, W2sw, t2, N, HID, nullptr);
  k_init_acc<<<(N * HID + thr - 1) / thr, thr, 0, stream>>>(t2, dinv, b2, emb, N * HID);
  k_edge_scatter<<<(E * 32 + thr - 1) / thr, thr, 0, stream>>>(t2, emb, src, dst, dinv, E);

  // 5) pooled linear: out = emb.reshape(M3, S*H) @ Wl + bl (reshape is a no-op)
  k_wmma_gemm<false><<<(M3 + 15) / 16, thr, 0, stream>>>(emb, Wlsw, out, M3, K3, bl);
}
